// MultiHeadAttentionLayer_35777077576232
// MI455X (gfx1250) — matrix operand order repaired and checked by reading, not compiled, not run on hardware
//
#include <hip/hip_runtime.h>
#include <hip/hip_bf16.h>

// Problem constants (from reference)
#define HIDN 512
#define NB   8
#define SEQ  1024
#define RNK  8
#define SSITERS 24

typedef __attribute__((ext_vector_type(16))) __bf16 v16bf;
typedef __attribute__((ext_vector_type(8)))  float  v8f;

// ---------------- WMMA helpers (16x16x32 bf16 -> f32) ----------------
// A-matrix 16x32 lane layout (ISA 7.12.2): lane=(r,h), element j -> K:
__device__ __forceinline__ int kidx_a(int j, int h) {
    return (j & 7) + (h << 3) + ((j >> 3) << 4);
}
// B-matrix 32x16: same K order as A (element j -> K = (j&7) + 8h + 16*(j>>3))
__device__ __forceinline__ int kidx_b(int j, int h) {
    return (j & 7) + (h << 3) + ((j >> 3) << 4);
}

// Load A tile: A[m][k] = src[m*ld + k], m = lane%16
__device__ __forceinline__ v16bf load_tile_a(const __bf16* src, int ld) {
    int lane = threadIdx.x & 31, r = lane & 15, h = lane >> 4;
    const __bf16* p = src + (size_t)r * ld;
    v16bf out;
#pragma unroll
    for (int j = 0; j < 16; ++j) out[j] = p[kidx_a(j, h)];
    return out;
}

// Load B tile where logical B[k][n] = srcT[n*ld + k]  (i.e. B = srcT^T)
__device__ __forceinline__ v16bf load_tile_bT(const __bf16* srcT, int ld) {
    int lane = threadIdx.x & 31, r = lane & 15, h = lane >> 4;
    const __bf16* p = srcT + (size_t)r * ld;
    v16bf out;
#pragma unroll
    for (int j = 0; j < 16; ++j) out[j] = p[kidx_b(j, h)];
    return out;
}

// Load B tile from f32 column block: B[k][n] = Y[k*ncols + n] for n<ncols else 0
__device__ __forceinline__ v16bf load_tile_b_f32cols(const float* Y, int ncols) {
    int lane = threadIdx.x & 31, r = lane & 15, h = lane >> 4;
    v16bf out;
#pragma unroll
    for (int j = 0; j < 16; ++j) {
        int k = kidx_b(j, h);
        float v = (r < ncols) ? Y[k * ncols + r] : 0.0f;
        out[j] = (__bf16)v;
    }
    return out;
}

__device__ __forceinline__ v8f wmma_bf16(v16bf a, v16bf b, v8f c) {
    return __builtin_amdgcn_wmma_f32_16x16x32_bf16(
        /*neg_a=*/false, a, /*neg_b=*/false, b,
        /*c_mod=*/(short)0, c, /*reuse_a=*/false, /*reuse_b=*/false);
}

// Pack 8 floats -> 8 bf16 -> one 16-byte store (dst must be 16B aligned)
__device__ __forceinline__ void store_bf16x8(__bf16* dst, const float* vals) {
    union { __bf16 h[8]; uint4 u; } pk;
#pragma unroll
    for (int v = 0; v < 8; ++v) pk.h[v] = (__bf16)vals[v];
    *(uint4*)dst = pk.u;
}

// ---------------- Kernels ----------------

__global__ void cvt_f32_to_bf16_kernel(const float* __restrict__ src,
                                       __bf16* __restrict__ dst, int n) {
    int i = blockIdx.x * blockDim.x + threadIdx.x;
    if (i < n) dst[i] = (__bf16)src[i];
}

// Xl = X @ W^T + bias.  Each wave computes a 16x32 slab (1 A tile, 2 B tiles).
// Writes transposed bf16 XT[b][o][s] (packed 16B stores); if WRITE_N also
// Xn[b][s][o].  grid (SEQ/16, HIDN/32/4, NB), block 128.
template <bool WRITE_N>
__global__ void proj_gemm_kernel(const __bf16* __restrict__ X,   // [NB,SEQ,HIDN]
                                 const __bf16* __restrict__ W,   // [HIDN,HIDN]
                                 const float*  __restrict__ bias,// [HIDN]
                                 __bf16* __restrict__ XT,        // [NB,HIDN,SEQ]
                                 __bf16* __restrict__ Xn) {      // [NB,SEQ,HIDN]
    int b = blockIdx.z;
    int m0 = blockIdx.x * 16;
    int wave = threadIdx.x >> 5;
    int n0 = (blockIdx.y * 4 + wave) * 32;
    const __bf16* Xb = X + (size_t)b * SEQ * HIDN;
    v8f acc0 = {}, acc1 = {};
    for (int k0 = 0; k0 < HIDN; k0 += 32) {
        v16bf a  = load_tile_a (Xb + (size_t)m0 * HIDN + k0, HIDN);
        v16bf b0 = load_tile_bT(W  + (size_t)n0 * HIDN + k0, HIDN);
        v16bf b1 = load_tile_bT(W  + (size_t)(n0 + 16) * HIDN + k0, HIDN);
        acc0 = wmma_bf16(a, b0, acc0);
        acc1 = wmma_bf16(a, b1, acc1);
    }
    int lane = threadIdx.x & 31, r = lane & 15, h = lane >> 4;
    __bf16* XTb = XT + (size_t)b * HIDN * SEQ;
    __bf16* Xnb = Xn + (size_t)b * SEQ * HIDN;
#pragma unroll
    for (int t = 0; t < 2; ++t) {
        v8f acc = t ? acc1 : acc0;
        int n = n0 + t * 16 + r;
        float bv = bias[n];
        float vals[8];
#pragma unroll
        for (int v = 0; v < 8; ++v) vals[v] = acc[v] + bv;
        // rows m0+8h .. m0+8h+7 are contiguous in the transposed layout
        store_bf16x8(XTb + (size_t)n * SEQ + m0 + 8 * h, vals);
        if (WRITE_N) {
#pragma unroll
            for (int v = 0; v < 8; ++v) {
                int m = m0 + v + 8 * h;
                Xnb[(size_t)m * HIDN + n] = (__bf16)vals[v];
            }
        }
    }
}

// G = T @ T^T, T = [HIDN, SEQ] bf16 rows.  16x32 slab per wave.
// grid (HIDN/16, HIDN/32/4, NB), block 128.
__global__ void gram_gemm_kernel(const __bf16* __restrict__ T,  // [NB,HIDN,SEQ]
                                 float* __restrict__ G) {       // [NB,HIDN,HIDN]
    int b = blockIdx.z;
    int i0 = blockIdx.x * 16;
    int wave = threadIdx.x >> 5;
    int j0 = (blockIdx.y * 4 + wave) * 32;
    const __bf16* Tb = T + (size_t)b * HIDN * SEQ;
    v8f acc0 = {}, acc1 = {};
    for (int k0 = 0; k0 < SEQ; k0 += 32) {
        v16bf a  = load_tile_a (Tb + (size_t)i0 * SEQ + k0, SEQ);
        v16bf b0 = load_tile_bT(Tb + (size_t)j0 * SEQ + k0, SEQ);
        v16bf b1 = load_tile_bT(Tb + (size_t)(j0 + 16) * SEQ + k0, SEQ);
        acc0 = wmma_bf16(a, b0, acc0);
        acc1 = wmma_bf16(a, b1, acc1);
    }
    int lane = threadIdx.x & 31, r = lane & 15, h = lane >> 4;
    float* Gb = G + (size_t)b * HIDN * HIDN;
#pragma unroll
    for (int t = 0; t < 2; ++t) {
        v8f acc = t ? acc1 : acc0;
        int j = j0 + t * 16 + r;
#pragma unroll
        for (int v = 0; v < 8; ++v) {
            int i = i0 + v + 8 * h;
            Gb[(size_t)i * HIDN + j] = acc[v];
        }
    }
}

// Subspace iteration: top-RNK eigenvectors of 512x512 PSD G.
// grid (2, NB), block 512.  blockIdx.x: 0 -> (Gq,Yq), 1 -> (Gv,Yv).
__global__ void subspace_kernel(const float* __restrict__ Gq,
                                const float* __restrict__ Gv,
                                float* __restrict__ Yq,
                                float* __restrict__ Yv) {
    int which = blockIdx.x, b = blockIdx.y;
    const float* G   = (which ? Gv : Gq) + (size_t)b * HIDN * HIDN;
    float*       Yo  = (which ? Yv : Yq) + (size_t)b * HIDN * RNK;
    __shared__ float Ys[HIDN][RNK];
    __shared__ float red[RNK][HIDN];
    int i = threadIdx.x;
    // deterministic pseudo-random init
#pragma unroll
    for (int c = 0; c < RNK; ++c) {
        unsigned u = (unsigned)i * 2654435761u ^ ((unsigned)c * 0x9E3779B9u) ^ 0x85EBCA6Bu;
        u = (u ^ (u >> 15)) * 0x2C1B3C6Du; u ^= u >> 12;
        Ys[i][c] = ((float)(u & 0xFFFFu) / 32768.0f) - 1.0f;
    }
    __syncthreads();
    float zr[RNK];
    const float* grow = G + (size_t)i * HIDN;
    for (int it = 0; it < SSITERS; ++it) {
        // Z = G @ Y (each thread owns row i)
#pragma unroll
        for (int c = 0; c < RNK; ++c) zr[c] = 0.0f;
        for (int jj = 0; jj < HIDN; ++jj) {
            if ((jj & 63) == 0) __builtin_prefetch(grow + jj + 64, 0, 0);
            float g = grow[jj];
#pragma unroll
            for (int c = 0; c < RNK; ++c) zr[c] += g * Ys[jj][c];
        }
        __syncthreads();  // all Ys reads done before rewrite below
        // Modified Gram-Schmidt across the 8 columns
        for (int c = 0; c < RNK; ++c) {
            for (int d = c; d < RNK; ++d) red[d][i] = zr[c] * zr[d];
            __syncthreads();
            for (int off = 256; off > 0; off >>= 1) {
                if (i < off)
                    for (int d = c; d < RNK; ++d) red[d][i] += red[d][i + off];
                __syncthreads();
            }
            float inv = rsqrtf(red[c][0] + 1e-20f);
            zr[c] *= inv;
#pragma unroll
            for (int d = c + 1; d < RNK; ++d) zr[d] -= (red[d][0] * inv) * zr[c];
            __syncthreads();  // red[*][0] consumed before next c rewrites
        }
#pragma unroll
        for (int c = 0; c < RNK; ++c) Ys[i][c] = zr[c];
        __syncthreads();
    }
#pragma unroll
    for (int c = 0; c < RNK; ++c) Yo[(size_t)i * RNK + c] = Ys[i][c];
}

// Out[M,32] = A[M,512] @ pad16(Y[512,8]); cols 8..31 zero-padded bf16.
// grid (M/16, 1, NB), block 32.
__global__ void lowrank_gemm_kernel(const __bf16* __restrict__ A, size_t aBatchStride,
                                    const float* __restrict__ Y,   // [NB,512,8]
                                    __bf16* __restrict__ Outp,     // [NB,M,32]
                                    int M) {
    int b = blockIdx.z;
    const __bf16* Ab = A + (size_t)b * aBatchStride;
    const float*  Yb = Y + (size_t)b * HIDN * RNK;
    __bf16*       Ob = Outp + (size_t)b * (size_t)M * 32;
    int m0 = blockIdx.x * 16;
    v8f acc = {};
    for (int k0 = 0; k0 < HIDN; k0 += 32) {
        v16bf a  = load_tile_a(Ab + (size_t)m0 * HIDN + k0, HIDN);
        v16bf bt = load_tile_b_f32cols(Yb + (size_t)k0 * RNK, RNK);
        acc = wmma_bf16(a, bt, acc);
    }
    int lane = threadIdx.x & 31, r = lane & 15, h = lane >> 4;
#pragma unroll
    for (int v = 0; v < 8; ++v) {
        int m = m0 + v + 8 * h;
        Ob[(size_t)m * 32 + r]      = (r < RNK) ? (__bf16)acc[v] : (__bf16)0.0f;
        Ob[(size_t)m * 32 + 16 + r] = (__bf16)0.0f;
    }
}

// out = Vc_pad[SEQ,32] @ W2_pad[HIDN,32]^T + bo.  grid (SEQ/16, HIDN/16/4, NB), block 128.
__global__ void final_gemm_kernel(const __bf16* __restrict__ Vc,  // [NB,SEQ,32]
                                  const __bf16* __restrict__ W2,  // [NB,HIDN,32]
                                  const float*  __restrict__ bo,
                                  float* __restrict__ out) {      // [NB,SEQ,HIDN]
    int b = blockIdx.z;
    int m0 = blockIdx.x * 16;
    int wave = threadIdx.x >> 5;
    int n0 = (blockIdx.y * 4 + wave) * 16;
    const __bf16* Vb = Vc + (size_t)b * SEQ * 32;
    const __bf16* Wb = W2 + (size_t)b * HIDN * 32;
    v16bf a  = load_tile_a (Vb + (size_t)m0 * 32, 32);
    v16bf bt = load_tile_bT(Wb + (size_t)n0 * 32, 32);
    v8f acc = {};
    acc = wmma_bf16(a, bt, acc);
    int lane = threadIdx.x & 31, r = lane & 15, h = lane >> 4;
    int n = n0 + r;
    float bias = bo[n];
    float* ob = out + (size_t)b * SEQ * HIDN;
#pragma unroll
    for (int v = 0; v < 8; ++v) {
        int m = m0 + v + 8 * h;
        ob[(size_t)m * HIDN + n] = acc[v] + bias;
    }
}

__global__ void ones_kernel(float* __restrict__ p, int n) {
    int i = blockIdx.x * blockDim.x + threadIdx.x;
    if (i < n) p[i] = 1.0f;
}

// ---------------- Launch ----------------
extern "C" void kernel_launch(void* const* d_in, const int* in_sizes, int n_in,
                              void* d_out, int out_size, void* d_ws, size_t ws_size,
                              hipStream_t stream) {
    const float* query = (const float*)d_in[0];
    const float* value = (const float*)d_in[2];
    const float* Wq    = (const float*)d_in[3];
    const float* bq    = (const float*)d_in[4];
    const float* Wv    = (const float*)d_in[7];
    const float* bv    = (const float*)d_in[8];
    const float* Wo    = (const float*)d_in[9];
    const float* bo    = (const float*)d_in[10];
    float* out = (float*)d_out;

    char* ws = (char*)d_ws;
    auto alloc = [&](size_t bytes) -> char* {
        char* p = ws;
        ws += (bytes + 255) & ~(size_t)255;
        return p;
    };
    const size_t actB = (size_t)NB * SEQ * HIDN;   // 4M elems
    const size_t wB   = (size_t)HIDN * HIDN;       // 256K elems
    __bf16* q_bf = (__bf16*)alloc(actB * 2);
    __bf16* v_bf = (__bf16*)alloc(actB * 2);
    __bf16* Wqb  = (__bf16*)alloc(wB * 2);
    __bf16* Wvb  = (__bf16*)alloc(wB * 2);
    __bf16* Wob  = (__bf16*)alloc(wB * 2);
    __bf16* QlT  = (__bf16*)alloc(actB * 2);       // [NB,HIDN,SEQ]
    __bf16* VlT  = (__bf16*)alloc(actB * 2);
    __bf16* Vln  = (__bf16*)alloc(actB * 2);       // [NB,SEQ,HIDN]
    float*  Gq   = (float*)alloc((size_t)NB * wB * 4);
    float*  Gv   = (float*)alloc((size_t)NB * wB * 4);
    float*  Yq   = (float*)alloc((size_t)NB * HIDN * RNK * 4);
    float*  Yv   = (float*)alloc((size_t)NB * HIDN * RNK * 4);
    __bf16* Vcp  = (__bf16*)alloc((size_t)NB * SEQ * 32 * 2);
    __bf16* W2p  = (__bf16*)alloc((size_t)NB * HIDN * 32 * 2);
    if ((size_t)(ws - (char*)d_ws) > ws_size) return;  // workspace too small

    // 1) convert inputs/weights to bf16
    {
        int n = (int)actB;
        cvt_f32_to_bf16_kernel<<<(n + 255) / 256, 256, 0, stream>>>(query, q_bf, n);
        cvt_f32_to_bf16_kernel<<<(n + 255) / 256, 256, 0, stream>>>(value, v_bf, n);
        int m = (int)wB;
        cvt_f32_to_bf16_kernel<<<(m + 255) / 256, 256, 0, stream>>>(Wq, Wqb, m);
        cvt_f32_to_bf16_kernel<<<(m + 255) / 256, 256, 0, stream>>>(Wv, Wvb, m);
        cvt_f32_to_bf16_kernel<<<(m + 255) / 256, 256, 0, stream>>>(Wo, Wob, m);
    }
    // 2) projections: Ql (transposed only), Vl (transposed + row-major)
    dim3 gp(SEQ / 16, HIDN / 32 / 4, NB);
    proj_gemm_kernel<false><<<gp, 128, 0, stream>>>(q_bf, Wqb, bq, QlT, Vln /*unused*/);
    proj_gemm_kernel<true ><<<gp, 128, 0, stream>>>(v_bf, Wvb, bv, VlT, Vln);
    // 3) Gram matrices (L2-resident afterwards)
    dim3 gg(HIDN / 16, HIDN / 32 / 4, NB);
    gram_gemm_kernel<<<gg, 128, 0, stream>>>(QlT, Gq);
    gram_gemm_kernel<<<gg, 128, 0, stream>>>(VlT, Gv);
    // 4) top-8 eigenvectors via subspace iteration
    subspace_kernel<<<dim3(2, NB), 512, 0, stream>>>(Gq, Gv, Yq, Yv);
    // 5) Vc = Vl @ V8(Vl)  and  W2 = Wo @ V8(Ql)   (N padded 8 -> 32)
    lowrank_gemm_kernel<<<dim3(SEQ / 16, 1, NB), 32, 0, stream>>>(
        Vln, (size_t)SEQ * HIDN, Yv, Vcp, SEQ);
    lowrank_gemm_kernel<<<dim3(HIDN / 16, 1, NB), 32, 0, stream>>>(
        Wob, (size_t)0, Yq, W2p, HIDN);
    // 6) out = Vc @ W2^T + bo
    final_gemm_kernel<<<dim3(SEQ / 16, HIDN / 16 / 4, NB), 128, 0, stream>>>(
        Vcp, W2p, bo, out);
    // 7) attention output = ones(64)
    ones_kernel<<<1, 64, 0, stream>>>(out + actB, NB * 8);
}
